// MemoLSTM_60601988547001
// MI455X (gfx1250) — compile-verified, matrix operand order checked by reading, not run on hardware
//
#include <hip/hip_runtime.h>
#include <hip/hip_bf16.h>

typedef __attribute__((ext_vector_type(16))) _Float16 v16h;
typedef __attribute__((ext_vector_type(8)))  _Float16 v8h;
typedef __attribute__((ext_vector_type(8)))  float    v8f;
typedef __attribute__((ext_vector_type(4)))  int      v4i;

#define DEV __device__ __forceinline__

DEV v8f wmma_step(v16h a, v16h b, v8f c) {
  return __builtin_amdgcn_wmma_f32_16x16x32_f16(false, a, false, b, (short)0, c, false, false);
}

DEV float sigm(float x) { return 1.0f / (1.0f + __expf(-x)); }

// zero a v8h via bitwise AND with a lane-uniform mask (branchless boundary handling)
DEV v8h and_mask(v8h v, int m) {
  union { v8h h; v4i i; } u;
  u.h = v;
  u.i &= m;
  return u.h;
}

// ---------------------------------------------------------------------------
// Generic WMMA GEMM: C[M,N] = act(A·B + bias). Compile-time layout flags:
// a(m,k) = TA ? A[k*M+m] : A[m*K+k] ; b(k,n) = TB ? B[n*K+k] : B[k*N+n]
// bias indexed by n. ACT: 0=none 1=relu 2=tanh. One wave per 16x16 tile.
// All index math is 32-bit; out-of-range rows/cols clamped for loads (their C
// entries are never stored); tail chunk is branchless (clamped k + cndmask);
// K loop double-buffered so loads overlap the WMMA.
// ---------------------------------------------------------------------------
template <int TA, int TB, int ACT>
__global__ void gemm16(const float* __restrict__ A, const float* __restrict__ Bm,
                       const float* __restrict__ bias, float* __restrict__ C,
                       int M, int N, int K,
                       long long sA, long long sB, long long sC)
{
  const int lane = threadIdx.x & 31;
  const int rc   = lane & 15;
  const int kh   = lane >> 4;
  const long long z = blockIdx.z;
  const float* Ap = A + z * sA;
  const float* Bp = Bm + z * sB;
  float*       Cp = C + z * sC;
  const int m0 = blockIdx.x * 16;
  const int n0 = blockIdx.y * 16;
  const int m  = m0 + rc;
  const int n  = n0 + rc;
  const int ml = (m < M) ? m : (M - 1);
  const int nl = (n < N) ? n : (N - 1);

  auto ldA = [&](int kb) {
    v16h a;
    if (TA == 0) {
      const float* ap = Ap + ml * K + kb + kh * 8;
#pragma unroll
      for (int i = 0; i < 8; ++i) { a[i] = (_Float16)ap[i]; a[i + 8] = (_Float16)ap[16 + i]; }
    } else {
#pragma unroll
      for (int i = 0; i < 16; ++i) {
        int k = kb + ((i >> 3) << 4) + kh * 8 + (i & 7);
        a[i] = (_Float16)Ap[k * M + ml];
      }
    }
    return a;
  };
  auto ldB = [&](int kb) {
    v16h b;
    if (TB == 1) {
      const float* bp = Bp + nl * K + kb + kh * 8;
#pragma unroll
      for (int i = 0; i < 8; ++i) { b[i] = (_Float16)bp[i]; b[i + 8] = (_Float16)bp[16 + i]; }
    } else {
#pragma unroll
      for (int i = 0; i < 16; ++i) {
        int k = kb + ((i >> 3) << 4) + kh * 8 + (i & 7);
        b[i] = (_Float16)Bp[k * N + nl];
      }
    }
    return b;
  };

  v8f acc = {};
  const int kfull = K & ~31;
  if (kfull > 0) {  // pipelined full chunks
    v16h a0 = ldA(0), b0 = ldB(0);
    for (int kb = 32; kb < kfull; kb += 32) {
      v16h a1 = ldA(kb), b1 = ldB(kb);
      acc = wmma_step(a0, b0, acc);
      a0 = a1; b0 = b1;
    }
    acc = wmma_step(a0, b0, acc);
  }
  if (kfull < K) {  // branchless tail: clamp k for address, cndmask the value
    v16h a, b;
#pragma unroll
    for (int i = 0; i < 16; ++i) {
      int k = kfull + ((i >> 3) << 4) + kh * 8 + (i & 7);
      bool ok = k < K;
      int kc = ok ? k : (K - 1);
      float av = TA ? Ap[kc * M + ml] : Ap[ml * K + kc];
      float bv = TB ? Bp[nl * K + kc] : Bp[kc * N + nl];
      a[i] = (_Float16)(ok ? av : 0.f);
      b[i] = (_Float16)(ok ? bv : 0.f);
    }
    acc = wmma_step(a, b, acc);
  }

  if (n < N) {
    float bv = bias ? bias[n] : 0.f;
    if (m0 + 16 <= M) {  // uniform full-tile fast path
#pragma unroll
      for (int j = 0; j < 8; ++j) {
        int mm = m0 + j + kh * 8;
        float v = acc[j] + bv;
        if (ACT == 1) v = fmaxf(v, 0.f);
        else if (ACT == 2) v = tanhf(v);
        Cp[mm * N + n] = v;
      }
    } else {
#pragma unroll
      for (int j = 0; j < 8; ++j) {
        int mm = m0 + j + kh * 8;
        if (mm < M) {
          float v = acc[j] + bv;
          if (ACT == 1) v = fmaxf(v, 0.f);
          else if (ACT == 2) v = tanhf(v);
          Cp[mm * N + n] = v;
        }
      }
    }
  }
}

// ---------------------------------------------------------------------------
// 3x3 SAME conv, NHWC f16 activations, weights pre-packed as [9][Cout][Cin] f16.
// Cin is ALWAYS a multiple of 32 (layer 1 zero-padded to 32). 9 shift passes;
// each pass is a K=Cin GEMM with both operands contiguous in K (16B-aligned
// v8h loads). Boundary taps branchless: clamped address + AND-mask. K loop
// double-buffered. Output NHWC f16, fused bias+ReLU.
// ---------------------------------------------------------------------------
__global__ void conv3x3_nhwc(const _Float16* __restrict__ I, const _Float16* __restrict__ Wp,
                             const float* __restrict__ bias, _Float16* __restrict__ O,
                             int Cin, int Cout, int H, int W)
{
  const int lane = threadIdx.x & 31;
  const int rc   = lane & 15;
  const int kh   = lane >> 4;
  const int HW   = H * W;
  const int m0   = blockIdx.x * 16;
  const int NT   = 8 * HW;
  const int n    = blockIdx.y * 16 + rc;
  const bool nok = (n < NT);
  const int nl   = nok ? n : (NT - 1);
  const int b = nl / HW;
  const int r = nl % HW;
  const int y = r / W, x = r - y * W;
  const int m = m0 + rc;  // always < Cout

  v8f acc = {};
#pragma unroll
  for (int p = 0; p < 9; ++p) {
    const int ky = p / 3, kx = p - ky * 3;
    const int iy = y + ky - 1, ix = x + kx - 1;
    const bool vld = (iy >= 0) && (iy < H) && (ix >= 0) && (ix < W);
    const int bm = vld ? -1 : 0;
    const _Float16* Bp = I + ((b * H + (vld ? iy : 0)) * W + (vld ? ix : 0)) * Cin + kh * 8;
    const _Float16* Aw = Wp + (p * Cout + m) * Cin + kh * 8;

    auto ldA = [&](int kb) {
      v8h lo = *(const v8h*)(Aw + kb);
      v8h hi = *(const v8h*)(Aw + kb + 16);
      v16h a;
#pragma unroll
      for (int i = 0; i < 8; ++i) { a[i] = lo[i]; a[i + 8] = hi[i]; }
      return a;
    };
    auto ldB = [&](int kb) {
      v8h lo = and_mask(*(const v8h*)(Bp + kb), bm);
      v8h hi = and_mask(*(const v8h*)(Bp + kb + 16), bm);
      v16h bb;
#pragma unroll
      for (int i = 0; i < 8; ++i) { bb[i] = lo[i]; bb[i + 8] = hi[i]; }
      return bb;
    };

    v16h a0 = ldA(0), b0 = ldB(0);
    for (int kb = 32; kb < Cin; kb += 32) {
      v16h a1 = ldA(kb), b1 = ldB(kb);
      acc = wmma_step(a0, b0, acc);
      a0 = a1; b0 = b1;
    }
    acc = wmma_step(a0, b0, acc);
  }

  if (nok) {
    const int co0 = m0 + 8 * kh;
    v8h ov;
#pragma unroll
    for (int j = 0; j < 8; ++j)
      ov[j] = (_Float16)fmaxf(acc[j] + bias[co0 + j], 0.f);
    *(v8h*)(O + nl * Cout + co0) = ov;
  }
}

// repack W[co][Cs][3][3] f32 -> Wp[p][co][Cd] f16, zero-padding Cs..Cd-1
__global__ void wrepack_k(const float* __restrict__ W, _Float16* __restrict__ Wp,
                          int Cout, int Cs, int Cd)
{
  int idx = blockIdx.x * 256 + threadIdx.x;
  int tot = Cout * Cd * 9;
  if (idx >= tot) return;
  int ci = idx % Cd;
  int r  = idx / Cd;
  int co = r % Cout;
  int p  = r / Cout;
  Wp[idx] = (ci < Cs) ? (_Float16)W[(co * Cs + ci) * 9 + p] : (_Float16)0.f;
}

// img NCHW f32 -> NHWC f16, channel-padded to 32 (c>=3 -> 0)
__global__ void img2nhwc_k(const float* __restrict__ I, _Float16* __restrict__ O)
{
  int idx = blockIdx.x * 256 + threadIdx.x;
  if (idx >= 8 * 224 * 224 * 32) return;
  int c = idx & 31;
  int r = idx >> 5;
  int x = r % 224; r /= 224;
  int y = r % 224;
  int b = r / 224;
  float v = (c < 3) ? I[((b * 3 + c) * 224 + y) * 224 + x] : 0.f;
  O[idx] = (_Float16)v;
}

// NHWC f16 2x2 maxpool
__global__ void maxpool2h(const _Float16* __restrict__ I, _Float16* __restrict__ O,
                          int C, int Hi, int Wi)
{
  int Ho = Hi >> 1, Wo = Wi >> 1;
  int idx = blockIdx.x * blockDim.x + threadIdx.x;
  int tot = 8 * Ho * Wo * C;
  if (idx >= tot) return;
  int c = idx % C;
  int r = idx / C;
  int wo = r % Wo; r /= Wo;
  int ho = r % Ho;
  int b  = r / Ho;
  const _Float16* p = I + ((b * Hi + ho * 2) * Wi + wo * 2) * C + c;
  int rs = Wi * C;
  float v = fmaxf(fmaxf((float)p[0], (float)p[C]), fmaxf((float)p[rs], (float)p[rs + C]));
  O[idx] = (_Float16)v;
}

__global__ void h2f_k(const _Float16* __restrict__ I, float* __restrict__ O, int n)
{
  int idx = blockIdx.x * 256 + threadIdx.x;
  if (idx < n) O[idx] = (float)I[idx];
}

__global__ void zero_k(float* p, int n) {
  int i = blockIdx.x * blockDim.x + threadIdx.x;
  if (i < n) p[i] = 0.f;
}

// xe[(s*8+b)*200+e] = emb[sents[b*64+s]*200+e]
__global__ void embed_k(const int* __restrict__ sents, const float* __restrict__ emb,
                        float* __restrict__ xe)
{
  int idx = blockIdx.x * 256 + threadIdx.x;
  if (idx >= 64 * 8 * 200) return;
  int e = idx % 200;
  int r = idx / 200;
  int b = r % 8;
  int s = r / 8;
  xe[idx] = emb[(long long)sents[b * 64 + s] * 200 + e];
}

// One LSTM step pointwise: gates i,f,g,o from pre[t]+hh+bhh; writes h,c and out.
__global__ void lstm_gate(const float* __restrict__ pre, const float* __restrict__ hh,
                          const float* __restrict__ bhh, float* __restrict__ h,
                          float* __restrict__ c, float* __restrict__ out,
                          int t, int off)
{
  int idx = blockIdx.x * blockDim.x + threadIdx.x;
  if (idx >= 8 * 256) return;
  int b = idx >> 8, j = idx & 255;
  const float* pt = pre + (t * 8 + b) * 1024;
  const float* hb = hh + b * 1024;
  float ig = sigm(pt[j]       + hb[j]       + bhh[j]);
  float fg = sigm(pt[256 + j] + hb[256 + j] + bhh[256 + j]);
  float gg = tanhf(pt[512 + j] + hb[512 + j] + bhh[512 + j]);
  float og = sigm(pt[768 + j] + hb[768 + j] + bhh[768 + j]);
  float cv = fg * c[idx] + ig * gg;
  c[idx] = cv;
  float hv = og * tanhf(cv);
  h[idx] = hv;
  out[(b * 64 + t) * 512 + off + j] = hv;
}

// softmax over s of X[(b*L+s)*ld + r] -> Y[(b*R+r)*L + s]. Block per (b,r), 64 thr.
__global__ void softmax_rc(const float* __restrict__ X, float* __restrict__ Y,
                           int R, int L, int ld)
{
  __shared__ float red[64];
  int q = blockIdx.x;
  int b = q / R, r = q - b * R;
  int t = threadIdx.x;
  float v = (t < L) ? X[(b * L + t) * ld + r] : -3.0e38f;
  red[t] = v; __syncthreads();
  for (int s = 32; s > 0; s >>= 1) { if (t < s) red[t] = fmaxf(red[t], red[t + s]); __syncthreads(); }
  float mx = red[0]; __syncthreads();
  float ev = (t < L) ? __expf(v - mx) : 0.f;
  red[t] = ev; __syncthreads();
  for (int s = 32; s > 0; s >>= 1) { if (t < s) red[t] += red[t + s]; __syncthreads(); }
  if (t < L) Y[q * L + t] = ev / red[0];
}

// alpha[b,s,k] = aff[b,s,k] / sum_s aff[b,s,k]. Block per (b,k), 64 threads.
__global__ void alpha_k(const float* __restrict__ aff, float* __restrict__ alpha)
{
  __shared__ float red[64];
  int q = blockIdx.x;
  int b = q / 49, k = q - b * 49;
  int s = threadIdx.x;
  float v = aff[(b * 64 + s) * 49 + k];
  red[s] = v; __syncthreads();
  for (int st = 32; st > 0; st >>= 1) { if (s < st) red[s] += red[s + st]; __syncthreads(); }
  alpha[(b * 64 + s) * 49 + k] = v / red[0];
}

// cosine gate: imf[q,:] = a_h[q,:] * (1 - cos(img,a_h)), q = b*49+k.
__global__ void gates_k(const float* __restrict__ vggoF, const float* __restrict__ a_h,
                        float* __restrict__ imf)
{
  __shared__ float rd0[256], rd1[256], rd2[256];
  int q = blockIdx.x;
  int t = threadIdx.x;
  const float* iv = vggoF + q * 512;
  const float* av = a_h + q * 512;
  float dot = 0.f, si = 0.f, sa = 0.f;
  for (int d = t; d < 512; d += 256) {
    float a = iv[d], b = av[d];
    dot += a * b; si += a * a; sa += b * b;
  }
  rd0[t] = dot; rd1[t] = si; rd2[t] = sa; __syncthreads();
  for (int s = 128; s > 0; s >>= 1) {
    if (t < s) { rd0[t] += rd0[t + s]; rd1[t] += rd1[t + s]; rd2[t] += rd2[t + s]; }
    __syncthreads();
  }
  float na = fmaxf(sqrtf(rd1[0]), 1e-6f);
  float nb = fmaxf(sqrtf(rd2[0]), 1e-6f);
  float g = 1.0f - rd0[0] / (na * nb);
  for (int d = t; d < 512; d += 256)
    imf[q * 512 + d] = av[d] * g;
}

// tmat[(b*2+m)*512+d] = m ? s_image : s_text
__global__ void tmat_k(const float* __restrict__ st, const float* __restrict__ si,
                       float* __restrict__ tm)
{
  int idx = blockIdx.x * 256 + threadIdx.x;
  if (idx >= 8192) return;
  int d = idx & 511;
  int r = idx >> 9;
  int m = r & 1, b = r >> 1;
  tm[idx] = m ? si[b * 512 + d] : st[b * 512 + d];
}

// wei[b,d,m] = mm[b,d,m] * (softmax_m(sfin[b,:]) + 1)
__global__ void wei_k(const float* __restrict__ st, const float* __restrict__ si,
                      const float* __restrict__ sfin, float* __restrict__ wei)
{
  int idx = blockIdx.x * 256 + threadIdx.x;
  if (idx >= 4096) return;
  int b = idx >> 9;
  float e0 = sfin[b * 2 + 0], e1 = sfin[b * 2 + 1];
  float mx = fmaxf(e0, e1);
  float x0 = __expf(e0 - mx), x1 = __expf(e1 - mx);
  float inv = 1.0f / (x0 + x1);
  wei[idx * 2 + 0] = st[idx] * (x0 * inv + 1.0f);
  wei[idx * 2 + 1] = si[idx] * (x1 * inv + 1.0f);
}

__global__ void pfsm_k(const float* __restrict__ pin, float* __restrict__ pf)
{
  int idx = blockIdx.x * 256 + threadIdx.x;
  if (idx >= 4096) return;
  float e0 = pin[idx * 2], e1 = pin[idx * 2 + 1];
  float mx = fmaxf(e0, e1);
  float x0 = __expf(e0 - mx), x1 = __expf(e1 - mx);
  float inv = 1.0f / (x0 + x1);
  pf[idx * 2]     = x0 * inv;
  pf[idx * 2 + 1] = x1 * inv;
}

// gamma[b,m] = sum_d Wf[b,d] * pf[b,d,m]. Block per (b,m)=16, 256 threads.
__global__ void gamma_k(const float* __restrict__ Wf, const float* __restrict__ pf,
                        float* __restrict__ gamma)
{
  __shared__ float red[256];
  int b = blockIdx.x >> 1, m = blockIdx.x & 1, t = threadIdx.x;
  float acc = 0.f;
  for (int d = t; d < 512; d += 256)
    acc += Wf[b * 512 + d] * pf[(b * 512 + d) * 2 + m];
  red[t] = acc; __syncthreads();
  for (int s = 128; s > 0; s >>= 1) { if (t < s) red[t] += red[t + s]; __syncthreads(); }
  if (t == 0) gamma[blockIdx.x] = red[0];
}

__global__ void final_k(const float* __restrict__ wei, const float* __restrict__ gamma,
                        float* __restrict__ out)
{
  int idx = blockIdx.x * 256 + threadIdx.x;
  if (idx >= 4096) return;
  int b = idx >> 9;
  out[idx] = wei[idx * 2] * gamma[b * 2] + wei[idx * 2 + 1] * gamma[b * 2 + 1];
}

// ---------------------------------------------------------------------------
extern "C" void kernel_launch(void* const* d_in, const int* in_sizes, int n_in,
                              void* d_out, int out_size, void* d_ws, size_t ws_size,
                              hipStream_t stream)
{
  (void)in_sizes; (void)n_in; (void)out_size; (void)ws_size;
  const float* img   = (const float*)d_in[0];
  const int*   sents = (const int*)d_in[1];
  const float* emb   = (const float*)d_in[2];
  const float* Wih_f = (const float*)d_in[3];
  const float* Whh_f = (const float*)d_in[4];
  const float* bih_f = (const float*)d_in[5];
  const float* bhh_f = (const float*)d_in[6];
  const float* Wih_b = (const float*)d_in[7];
  const float* Whh_b = (const float*)d_in[8];
  const float* bih_b = (const float*)d_in[9];
  const float* bhh_b = (const float*)d_in[10];
  // vgg params at 11..42 (16 x (W,b))
  const float* Ws1 = (const float*)d_in[43];
  const float* bs1 = (const float*)d_in[44];
  const float* Ws2 = (const float*)d_in[45];
  const float* bs2 = (const float*)d_in[46];
  const float* l1W = (const float*)d_in[47];
  const float* l1b = (const float*)d_in[48];
  const float* Wb  = (const float*)d_in[49];
  const float* WF  = (const float*)d_in[50];
  const float* Wf  = (const float*)d_in[51];
  const float* a1W = (const float*)d_in[52];
  const float* a1b = (const float*)d_in[53];
  const float* a2W = (const float*)d_in[54];
  const float* a2b = (const float*)d_in[55];
  const float* a3W = (const float*)d_in[56];
  const float* a3b = (const float*)d_in[57];
  const float* a4W = (const float*)d_in[58];
  const float* a4b = (const float*)d_in[59];
  float* outp = (float*)d_out;

  // ---- workspace layout (byte allocator, 256B aligned) ----
  char* base = (char*)d_ws;
  size_t off = 0;
  auto allocB = [&](size_t bytes) -> void* {
    void* p = base + off; off += (bytes + 255) & ~(size_t)255; return p;
  };
  _Float16* hp    = (_Float16*)allocB((size_t)8 * 224 * 224 * 64 * 2);  // NHWC ping
  _Float16* hq    = (_Float16*)allocB((size_t)8 * 224 * 224 * 64 * 2);  // NHWC pong
  _Float16* imgh  = (_Float16*)allocB((size_t)8 * 224 * 224 * 32 * 2);  // padded to C=32
  _Float16* wpk   = (_Float16*)allocB((size_t)9 * 512 * 512 * 2);
  float* vggoF  = (float*)allocB((size_t)8 * 49 * 512 * 4);
  float* xe     = (float*)allocB((size_t)64 * 8 * 200 * 4);
  float* pre_f  = (float*)allocB((size_t)64 * 8 * 1024 * 4);
  float* pre_b  = (float*)allocB((size_t)64 * 8 * 1024 * 4);
  float* hc     = (float*)allocB((size_t)4 * 2048 * 4);
  float* hF = hc, *cF = hc + 2048, *hB = hc + 4096, *cB = hc + 6144;
  float* hhF    = (float*)allocB((size_t)8 * 1024 * 4);
  float* hhB    = (float*)allocB((size_t)8 * 1024 * 4);
  float* outL   = (float*)allocB((size_t)8 * 64 * 512 * 4);
  float* a1buf  = (float*)allocB((size_t)512 * 350 * 4);
  float* a2buf  = (float*)allocB((size_t)512 * 30 * 4);
  float* attn   = (float*)allocB((size_t)8 * 30 * 64 * 4);
  float* hidden = (float*)allocB((size_t)8 * 30 * 512 * 4);
  float* tmpTE  = (float*)allocB((size_t)8 * 64 * 512 * 4);
  float* aff    = (float*)allocB((size_t)8 * 64 * 49 * 4);
  float* alpha  = (float*)allocB((size_t)8 * 64 * 49 * 4);
  float* a_h    = (float*)allocB((size_t)8 * 49 * 512 * 4);
  float* imf    = (float*)allocB((size_t)8 * 49 * 512 * 4);
  float* attn2  = (float*)allocB((size_t)8 * 30 * 49 * 4);
  float* aimf   = (float*)allocB((size_t)8 * 30 * 512 * 4);
  float* s_text = (float*)allocB((size_t)8 * 512 * 4);
  float* s_img  = (float*)allocB((size_t)8 * 512 * 4);
  float* tmat   = (float*)allocB((size_t)16 * 512 * 4);
  float* u1     = (float*)allocB((size_t)16 * 256 * 4);
  float* u2     = (float*)allocB((size_t)16 * 64 * 4);
  float* u3     = (float*)allocB((size_t)16 * 8 * 4);
  float* sfin   = (float*)allocB((size_t)16 * 4);
  float* wei    = (float*)allocB((size_t)8 * 512 * 2 * 4);
  float* pfin   = (float*)allocB((size_t)8 * 512 * 2 * 4);
  float* pf     = (float*)allocB((size_t)8 * 512 * 2 * 4);
  float* gamma  = (float*)allocB((size_t)16 * 4);

  // ---- text path: embedding + biLSTM ----
  embed_k<<<(64 * 8 * 200 + 255) / 256, 256, 0, stream>>>(sents, emb, xe);
  gemm16<0, 1, 0><<<dim3(32, 64, 1), 32, 0, stream>>>(xe, Wih_f, bih_f, pre_f, 512, 1024, 200, 0, 0, 0);
  gemm16<0, 1, 0><<<dim3(32, 64, 1), 32, 0, stream>>>(xe, Wih_b, bih_b, pre_b, 512, 1024, 200, 0, 0, 0);
  zero_k<<<(8192 + 255) / 256, 256, 0, stream>>>(hc, 8192);
  for (int t = 0; t < 64; ++t) {
    gemm16<0, 1, 0><<<dim3(1, 64, 1), 32, 0, stream>>>(hF, Whh_f, nullptr, hhF, 8, 1024, 256, 0, 0, 0);
    lstm_gate<<<8, 256, 0, stream>>>(pre_f, hhF, bhh_f, hF, cF, outL, t, 0);
  }
  for (int t = 63; t >= 0; --t) {
    gemm16<0, 1, 0><<<dim3(1, 64, 1), 32, 0, stream>>>(hB, Whh_b, nullptr, hhB, 8, 1024, 256, 0, 0, 0);
    lstm_gate<<<8, 256, 0, stream>>>(pre_b, hhB, bhh_b, hB, cB, outL, t, 256);
  }

  // ---- attention over text ----
  gemm16<0, 1, 2><<<dim3(32, 22, 1), 32, 0, stream>>>(outL, Ws1, bs1, a1buf, 512, 350, 512, 0, 0, 0);
  gemm16<0, 1, 0><<<dim3(32, 2, 1), 32, 0, stream>>>(a1buf, Ws2, bs2, a2buf, 512, 30, 350, 0, 0, 0);
  softmax_rc<<<240, 64, 0, stream>>>(a2buf, attn, 30, 64, 30);
  gemm16<0, 0, 0><<<dim3(2, 32, 8), 32, 0, stream>>>(attn, outL, nullptr, hidden, 30, 512, 64,
                                                     30 * 64, 64 * 512, 30 * 512);

  // ---- VGG (NHWC f16, implicit-GEMM WMMA, 9-pass, layer1 padded to Cin=32) ----
  {
    img2nhwc_k<<<(8 * 224 * 224 * 32 + 255) / 256, 256, 0, stream>>>(img, imgh);
    const int cfg[21] = {64,64,0,128,128,0,256,256,256,256,0,512,512,512,512,0,512,512,512,512,0};
    const _Float16* cur = imgh;
    int H = 224, W = 224, cin = 3, cpad = 32, wi = 11;
    for (int i = 0; i < 21; ++i) {
      if (cfg[i] == 0) {
        _Float16* dst = (cur == hp) ? hq : hp;
        int tot = 8 * (H / 2) * (W / 2) * cin;
        maxpool2h<<<(tot + 255) / 256, 256, 0, stream>>>(cur, dst, cin, H, W);
        cur = dst; H /= 2; W /= 2;
      } else {
        int cout = cfg[i];
        wrepack_k<<<(cout * cpad * 9 + 255) / 256, 256, 0, stream>>>(
            (const float*)d_in[wi], wpk, cout, cin, cpad);
        _Float16* dst = (cur == hp) ? hq : hp;
        int NT = 8 * H * W;
        dim3 g(cout / 16, (unsigned)((NT + 15) / 16), 1);
        conv3x3_nhwc<<<g, 32, 0, stream>>>(cur, wpk, (const float*)d_in[wi + 1], dst, cpad, cout, H, W);
        cur = dst; cin = cout; cpad = cout; wi += 2;
      }
    }
    // vgg out NHWC [8,7,7,512] == [8,49,512]
    h2f_k<<<(8 * 49 * 512 + 255) / 256, 256, 0, stream>>>(cur, vggoF, 8 * 49 * 512);
  }

  // ---- image_filter ----
  gemm16<0, 0, 0><<<dim3(4, 32, 8), 32, 0, stream>>>(outL, Wb, nullptr, tmpTE, 64, 512, 512,
                                                     64 * 512, 512 * 512, 64 * 512);
  gemm16<0, 1, 0><<<dim3(4, 4, 8), 32, 0, stream>>>(tmpTE, vggoF, nullptr, aff, 64, 49, 512,
                                                    64 * 512, 49 * 512, 64 * 49);
  alpha_k<<<392, 64, 0, stream>>>(aff, alpha);
  gemm16<1, 0, 0><<<dim3(4, 32, 8), 32, 0, stream>>>(alpha, outL, nullptr, a_h, 49, 512, 64,
                                                     64 * 49, 64 * 512, 49 * 512);
  gates_k<<<392, 256, 0, stream>>>(vggoF, a_h, imf);

  // ---- attention over image features ----
  gemm16<0, 1, 2><<<dim3(25, 22, 1), 32, 0, stream>>>(imf, Ws1, bs1, a1buf, 392, 350, 512, 0, 0, 0);
  gemm16<0, 1, 0><<<dim3(25, 2, 1), 32, 0, stream>>>(a1buf, Ws2, bs2, a2buf, 392, 30, 350, 0, 0, 0);
  softmax_rc<<<240, 64, 0, stream>>>(a2buf, attn2, 30, 49, 30);
  gemm16<0, 0, 0><<<dim3(2, 32, 8), 32, 0, stream>>>(attn2, imf, nullptr, aimf, 30, 512, 49,
                                                     30 * 49, 49 * 512, 30 * 512);

  // ---- fusion heads ----
  gemm16<0, 1, 0><<<dim3(1, 32, 1), 32, 0, stream>>>(hidden, l1W, l1b, s_text, 8, 512, 15360, 0, 0, 0);
  gemm16<0, 1, 0><<<dim3(1, 32, 1), 32, 0, stream>>>(aimf, l1W, l1b, s_img, 8, 512, 15360, 0, 0, 0);

  // ---- ATMF ----
  tmat_k<<<32, 256, 0, stream>>>(s_text, s_img, tmat);
  gemm16<0, 1, 0><<<dim3(1, 16, 1), 32, 0, stream>>>(tmat, a1W, a1b, u1, 16, 256, 512, 0, 0, 0);
  gemm16<0, 1, 0><<<dim3(1, 4, 1), 32, 0, stream>>>(u1, a2W, a2b, u2, 16, 64, 256, 0, 0, 0);
  gemm16<0, 1, 0><<<dim3(1, 1, 1), 32, 0, stream>>>(u2, a3W, a3b, u3, 16, 8, 64, 0, 0, 0);
  gemm16<0, 1, 0><<<dim3(1, 1, 1), 32, 0, stream>>>(u3, a4W, a4b, sfin, 16, 1, 8, 0, 0, 0);
  wei_k<<<16, 256, 0, stream>>>(s_text, s_img, sfin, wei);
  gemm16<0, 0, 2><<<dim3(32, 1, 8), 32, 0, stream>>>(WF, wei, nullptr, pfin, 512, 2, 512,
                                                     512 * 512, 512 * 2, 512 * 2);
  pfsm_k<<<16, 256, 0, stream>>>(pfin, pf);
  gamma_k<<<16, 256, 0, stream>>>(Wf, pf, gamma);
  final_k<<<16, 256, 0, stream>>>(wei, gamma, outp);
}
